// TinyRNN_13623636263551
// MI455X (gfx1250) — compile-verified, matrix operand order checked by reading, not run on hardware
//
#include <hip/hip_runtime.h>
#include <hip/hip_bf16.h>
#include <math.h>

// ---------------------------------------------------------------------------
// TinyRNN (2-layer tanh RNN) for MI455X / gfx1250, wave32 + WMMA bf16.
//   B=64, T=2048, I=H=512, L=2
// Strategy:
//   1) Apre = x @ W_ih + (b_ih+b_hh)         -- big WMMA GEMM (M=131072)
//   2) persistent scan: h = tanh(Apre[t] + h @ W_hh)  (8 WGs, W_hh^T in LDS,
//      device-scope atomic barrier per timestep)
//   3) repeat for layer 1; layer-1 scan writes output + final_h.
// ---------------------------------------------------------------------------

typedef __attribute__((ext_vector_type(16))) __bf16 v16bf;
typedef __attribute__((ext_vector_type(8)))  float  v8f;

#define DEV_INLINE __device__ __forceinline__

static constexpr int B = 64;
static constexpr int T = 2048;
static constexpr int I = 512;
static constexpr int H = 512;
static constexpr int K = 512;        // reduction length (I == H)
static constexpr int NWG_SCAN = 8;   // persistent workgroups in scan

// 16x32 bf16 fragment (A or B^T operand of v_wmma_f32_16x16x32_bf16)
union Frag {
  v16bf v;
  __bf16 e[16];
  uint4  q[2];
};

// Row-major [*, ld] bf16 source; per-lane addressing per ISA 7.12.2:
//   lane<16 : row = base+lane,    K octets {0..7},{16..23}   of this k-chunk
//   lane>=16: row = base+lane-16, K octets {8..15},{24..31}
DEV_INLINE Frag load_frag_bf16(const __bf16* __restrict__ base, int row,
                               int ld, int kbase, int lane) {
  Frag f;
  const int r    = row + (lane & 15);
  const int koff = kbase + ((lane >> 4) << 3);
  const __bf16* p = base + (size_t)r * ld + koff;
  f.q[0] = *(const uint4*)(p);        // 8 bf16
  f.q[1] = *(const uint4*)(p + 16);   // 8 bf16 (k+16)
  return f;
}

// A-fragment straight from fp32 x[B][T][I]; logical GEMM row m = t*B + b.
DEV_INLINE Frag load_frag_x(const float* __restrict__ x, int m, int kbase,
                            int lane) {
  const int mm = m + (lane & 15);
  const int b  = mm & (B - 1);
  const int t  = mm >> 6;              // B == 64
  const float* p = x + ((size_t)b * T + t) * I + kbase + ((lane >> 4) << 3);
  float4 f0 = *(const float4*)(p);
  float4 f1 = *(const float4*)(p + 4);
  float4 f2 = *(const float4*)(p + 16);
  float4 f3 = *(const float4*)(p + 20);
  float tmp[16] = {f0.x, f0.y, f0.z, f0.w, f1.x, f1.y, f1.z, f1.w,
                   f2.x, f2.y, f2.z, f2.w, f3.x, f3.y, f3.z, f3.w};
  Frag f;
#pragma unroll
  for (int i = 0; i < 16; ++i) f.e[i] = (__bf16)tmp[i];
  return f;
}

DEV_INLINE v8f wmma_bf16(const Frag& a, const Frag& b, v8f c) {
  return __builtin_amdgcn_wmma_f32_16x16x32_bf16(
      /*neg_a=*/false, a.v, /*neg_b=*/false, b.v,
      /*c_mod=*/(short)0, c, /*reuse_a=*/false, /*reuse_b=*/false);
}

// ---------------------------------------------------------------------------
// init: zero h0, zero barrier counters, fuse biases
// ---------------------------------------------------------------------------
__global__ __launch_bounds__(256) void init_kernel(
    const float* __restrict__ b_ih0, const float* __restrict__ b_hh0,
    const float* __restrict__ b_ih1, const float* __restrict__ b_hh1,
    __bf16* __restrict__ h_init, float* __restrict__ bsum0,
    float* __restrict__ bsum1, unsigned* __restrict__ bars) {
  const int i = blockIdx.x * blockDim.x + threadIdx.x;
  if (i < B * H) h_init[i] = (__bf16)0.0f;
  if (i < H) {
    bsum0[i] = b_ih0[i] + b_hh0[i];
    bsum1[i] = b_ih1[i] + b_hh1[i];
  }
  if (i < 2) bars[i] = 0u;
}

// ---------------------------------------------------------------------------
// transpose + convert all 4 weight matrices: wt[mat][n][k] = (bf16)W[k][n]
// ---------------------------------------------------------------------------
__global__ __launch_bounds__(256) void prep_weights_kernel(
    const float* __restrict__ w_ih0, const float* __restrict__ w_hh0,
    const float* __restrict__ w_ih1, const float* __restrict__ w_hh1,
    __bf16* __restrict__ wt) {
  const float* srcs[4] = {w_ih0, w_hh0, w_ih1, w_hh1};
  const int total = 4 * K * H;
  for (int idx = blockIdx.x * blockDim.x + threadIdx.x; idx < total;
       idx += gridDim.x * blockDim.x) {
    const int mat = idx >> 18;           // 2^18 = 512*512
    const int rem = idx & (K * H - 1);
    const int n = rem >> 9;
    const int k = rem & 511;
    wt[idx] = (__bf16)srcs[mat][(size_t)k * H + n];
  }
}

// ---------------------------------------------------------------------------
// GEMM: Out[m][n] = Asrc[m][:] . Wt[n][:] + bias[n]
//   M = gridDim.x*64 (m = t*B+b time-major), N = H = 512, K = 512.
//   Block = 128 threads (4 waves); wave w -> 16 rows; 4 N-tiles of 16.
// ---------------------------------------------------------------------------
template <bool XF32>
__global__ __launch_bounds__(128) void gemm_kernel(
    const void* __restrict__ Asrc, const __bf16* __restrict__ Wt,
    const float* __restrict__ bias, float* __restrict__ Out) {
  const int lane = threadIdx.x & 31;
  const int wave = threadIdx.x >> 5;
  const int m0 = blockIdx.x * 64 + wave * 16;
  const int n0 = blockIdx.y * 64;

  v8f acc[4] = {};
#pragma unroll
  for (int kb = 0; kb < K; kb += 32) {
    Frag a;
    if constexpr (XF32)
      a = load_frag_x((const float*)Asrc, m0, kb, lane);
    else
      a = load_frag_bf16((const __bf16*)Asrc, m0, K, kb, lane);
#pragma unroll
    for (int nt = 0; nt < 4; ++nt) {
      Frag b = load_frag_bf16(Wt, n0 + nt * 16, K, kb, lane);
      acc[nt] = wmma_bf16(a, b, acc[nt]);
    }
  }

  const int rbase = (lane >> 4) * 8;
  const int cl = lane & 15;
#pragma unroll
  for (int nt = 0; nt < 4; ++nt) {
    const int n = n0 + nt * 16 + cl;
    const float bv = bias[n];
#pragma unroll
    for (int r = 0; r < 8; ++r) {
      const int m = m0 + rbase + r;
      Out[(size_t)m * H + n] = acc[nt][r] + bv;
    }
  }
}

// ---------------------------------------------------------------------------
// Persistent recurrent scan:  h_t = tanh(Apre[t] + h_{t-1} @ W_hh)
//   8 WGs x 256 threads. WG g owns 64 output columns; W_hh^T slice in LDS.
//   Wave w: M-tile (w&3)*16, two N-tiles (w>>2)*32 + {0,16}. 16 tiles total.
//   Device-scope atomic barrier between timesteps.
// ---------------------------------------------------------------------------
__global__ __launch_bounds__(256) void scan_kernel(
    const float* __restrict__ Apre,   // [T*B][H], m = t*B + b, bias fused
    const __bf16* __restrict__ Wt,    // [H][K] = W_hh transposed, bf16
    const __bf16* __restrict__ h0,    // [B][H] zeros
    __bf16* __restrict__ hbuf,        // [.][B][H] bf16 state history/ping-pong
    unsigned idx_mask,                // 0x7FFFFFFF: full history, 1: ping-pong
    float* __restrict__ out,          // [B][T][H] fp32 or nullptr
    float* __restrict__ final_dst,    // [B][H] fp32
    unsigned* __restrict__ bar) {
  extern __shared__ char smem_raw[];
  __bf16* ldsW = (__bf16*)smem_raw;   // [64][512] bf16 = 64 KB

  const int tid = threadIdx.x;
  const int lane = tid & 31;
  const int wave = tid >> 5;
  const int g = blockIdx.x;
  const int n0 = g * 64;
  const int mr = (wave & 3) * 16;
  const int ncA = (wave >> 2) * 32;
  const int ncB = ncA + 16;

  {  // stage W_hh^T slice (rows n0..n0+63) into LDS, 128-bit chunks
    const uint4* src = (const uint4*)(Wt + (size_t)n0 * K);
    uint4* dst = (uint4*)ldsW;
    for (int c = tid; c < 64 * K / 8; c += 256) dst[c] = src[c];
  }
  __syncthreads();

  const int rbase = (lane >> 4) * 8;
  const int cl = lane & 15;

  for (int t = 0; t < T; ++t) {
    const __bf16* hsrc =
        (t == 0) ? h0 : hbuf + (size_t)((unsigned)(t - 1) & idx_mask) * (B * H);
    __bf16* hdst = hbuf + (size_t)((unsigned)t & idx_mask) * (B * H);

    v8f acc0 = {}, acc1 = {};
#pragma unroll
    for (int kb = 0; kb < K; kb += 32) {
      Frag a = load_frag_bf16(hsrc, mr, H, kb, lane);
      Frag b0 = load_frag_bf16(ldsW, ncA, K, kb, lane);
      Frag b1 = load_frag_bf16(ldsW, ncB, K, kb, lane);
      acc0 = wmma_bf16(a, b0, acc0);
      acc1 = wmma_bf16(a, b1, acc1);
    }

#pragma unroll
    for (int r = 0; r < 8; ++r) {
      const int b = mr + rbase + r;  // batch row
      const float* ap = Apre + ((size_t)t * B + b) * H + n0;
      {
        const int n = ncA + cl;
        const float v = tanhf(acc0[r] + ap[n]);
        hdst[(size_t)b * H + n0 + n] = (__bf16)v;
        if (out) out[(size_t)b * T * H + (size_t)t * H + n0 + n] = v;
        if (t == T - 1) final_dst[(size_t)b * H + n0 + n] = v;
      }
      {
        const int n = ncB + cl;
        const float v = tanhf(acc1[r] + ap[n]);
        hdst[(size_t)b * H + n0 + n] = (__bf16)v;
        if (out) out[(size_t)b * T * H + (size_t)t * H + n0 + n] = v;
        if (t == T - 1) final_dst[(size_t)b * H + n0 + n] = v;
      }
    }

    // ---- device-scope barrier between timesteps ----
    __threadfence();     // make this WG's h stores visible at device scope
    __syncthreads();
    if (tid == 0) {
      __hip_atomic_fetch_add(bar, 1u, __ATOMIC_RELEASE,
                             __HIP_MEMORY_SCOPE_AGENT);
      const unsigned target = (unsigned)(t + 1) * NWG_SCAN;
      while (__hip_atomic_load(bar, __ATOMIC_ACQUIRE,
                               __HIP_MEMORY_SCOPE_AGENT) < target) {
        __builtin_amdgcn_s_sleep(2);
      }
    }
    __syncthreads();
    __threadfence();     // invalidate stale lines before reading peers' h
  }
}

// ---------------------------------------------------------------------------
extern "C" void kernel_launch(void* const* d_in, const int* in_sizes, int n_in,
                              void* d_out, int out_size, void* d_ws,
                              size_t ws_size, hipStream_t stream) {
  (void)in_sizes; (void)n_in; (void)out_size; (void)ws_size;
  const float* x     = (const float*)d_in[0];
  const float* w_ih0 = (const float*)d_in[1];
  const float* w_hh0 = (const float*)d_in[2];
  const float* b_ih0 = (const float*)d_in[3];
  const float* b_hh0 = (const float*)d_in[4];
  const float* w_ih1 = (const float*)d_in[5];
  const float* w_hh1 = (const float*)d_in[6];
  const float* b_ih1 = (const float*)d_in[7];
  const float* b_hh1 = (const float*)d_in[8];
  float* out = (float*)d_out;

  char* ws = (char*)d_ws;
  size_t off = 0;
  auto alloc = [&](size_t bytes) {
    off = (off + 255) & ~(size_t)255;
    char* p = ws + off;
    off += bytes;
    return p;
  };
  __bf16*   wt     = (__bf16*)alloc(4ull * K * H * sizeof(__bf16));
  float*    bsum0  = (float*)alloc(H * sizeof(float));
  float*    bsum1  = (float*)alloc(H * sizeof(float));
  __bf16*   h_init = (__bf16*)alloc((size_t)B * H * sizeof(__bf16));
  __bf16*   hbuf1  = (__bf16*)alloc(2ull * B * H * sizeof(__bf16));
  unsigned* bars   = (unsigned*)alloc(256);
  float*    Apre   = (float*)alloc((size_t)T * B * H * sizeof(float));   // 256 MB
  __bf16*   H0     = (__bf16*)alloc((size_t)T * B * H * sizeof(__bf16)); // 128 MB

  const __bf16* wt_ih0 = wt + 0ull * K * H;
  const __bf16* wt_hh0 = wt + 1ull * K * H;
  const __bf16* wt_ih1 = wt + 2ull * K * H;
  const __bf16* wt_hh1 = wt + 3ull * K * H;

  init_kernel<<<128, 256, 0, stream>>>(b_ih0, b_hh0, b_ih1, b_hh1, h_init,
                                       bsum0, bsum1, bars);
  prep_weights_kernel<<<1024, 256, 0, stream>>>(w_ih0, w_hh0, w_ih1, w_hh1, wt);

  const dim3 ggrid(T * B / 64, H / 64);  // 2048 x 8
  const size_t scan_lds = 64ull * K * sizeof(__bf16);  // 64 KB

  // Layer 0
  gemm_kernel<true><<<ggrid, 128, 0, stream>>>((const void*)x, wt_ih0, bsum0,
                                               Apre);
  scan_kernel<<<NWG_SCAN, 256, scan_lds, stream>>>(
      Apre, wt_hh0, h_init, H0, 0x7FFFFFFFu, nullptr,
      out + (size_t)B * T * H, bars + 0);

  // Layer 1
  gemm_kernel<false><<<ggrid, 128, 0, stream>>>((const void*)H0, wt_ih1, bsum1,
                                                Apre);
  scan_kernel<<<NWG_SCAN, 256, scan_lds, stream>>>(
      Apre, wt_hh1, h_init, hbuf1, 1u, out,
      out + (size_t)B * T * H + (size_t)B * H, bars + 1);
}
